// GPTBlock_39522289058092
// MI455X (gfx1250) — matrix operand order repaired and checked by reading, not compiled, not run on hardware
//
#include <hip/hip_runtime.h>

// ---------------------------------------------------------------------------
// GPT block for MI455X (gfx1250, wave32).  bf16 WMMA, fp32 accum.
// Data movement: TDM tensor_load_to_lds (weights/k tiles) + per-lane
// global_load_async_to_lds_b128 (activations), double-buffered in LDS.
// ---------------------------------------------------------------------------

typedef unsigned short u16;
typedef __attribute__((ext_vector_type(16))) __bf16 v16bf;
typedef __attribute__((ext_vector_type(8)))  float  v8f;
typedef __attribute__((ext_vector_type(4)))  unsigned int u32x4;
typedef __attribute__((ext_vector_type(8)))  int i32x8;
typedef __attribute__((ext_vector_type(4)))  int i32x4;

// exact parameter types for the async-to-LDS builtin (from hipcc diagnostic):
typedef int i32x4v __attribute__((vector_size(16)));
typedef __attribute__((address_space(1))) i32x4v* as1_v4i;
typedef __attribute__((address_space(3))) i32x4v* as3_v4i;

#if __has_builtin(__builtin_amdgcn_tensor_load_to_lds)
#define HAVE_TDM 1
#else
#define HAVE_TDM 0
#endif

__device__ __forceinline__ u16 f2bf(float x) {
    unsigned u = __float_as_uint(x);
    u += 0x7fffu + ((u >> 16) & 1u);          // round-to-nearest-even
    return (u16)(u >> 16);
}

// ---- async copy: 16 bytes global -> LDS (per lane) ------------------------
__device__ __forceinline__ void cp_async16(const u16* g, u16* l) {
#if __has_builtin(__builtin_amdgcn_global_load_async_to_lds_b128)
    __builtin_amdgcn_global_load_async_to_lds_b128(
        (as1_v4i)(uintptr_t)g, (as3_v4i)(uintptr_t)l, 0, 0);
#else
    *(u32x4*)l = *(const u32x4*)g;            // sync fallback
#endif
}

__device__ __forceinline__ void wait_async0() {
#if __has_builtin(__builtin_amdgcn_s_wait_asynccnt)
    __builtin_amdgcn_s_wait_asynccnt(0);
#else
    asm volatile("s_wait_asynccnt 0x0" ::: "memory");
#endif
}

__device__ __forceinline__ void wait_tensor0() {
#if __has_builtin(__builtin_amdgcn_s_wait_tensorcnt)
    __builtin_amdgcn_s_wait_tensorcnt(0);
#else
    asm volatile("s_wait_tensorcnt 0x0" ::: "memory");
#endif
}

// ---- TDM: 2-D bf16 tile (tile_y rows x tile_x elems) global -> LDS --------
// D# per CDNA5 ISA ch.8: group0 {count=1, lds_addr, global_addr, type=2},
// group1 {data_size=2B, tensor dims, tile dims, dim0 stride}, groups2/3 = 0.
// This toolchain's builtin is the 6-arg form:
//   (uint32x4 g0, int32x8 g1, int32x4 g2, int32x4 g3, int32x8 pad, i32 cpol)
__device__ __forceinline__ void tdm_load_2d(unsigned lds_off, const void* gaddr,
                                            unsigned tile_x, unsigned tile_y,
                                            unsigned tensor_x, unsigned tensor_y,
                                            unsigned stride_x /*elements*/) {
#if HAVE_TDM
    unsigned long long ga = (unsigned long long)(uintptr_t)gaddr;
    u32x4 g0;
    g0.x = 1u;                                        // count=1, user mode
    g0.y = lds_off;                                   // lds_addr (bytes)
    g0.z = (unsigned)(ga & 0xFFFFFFFFu);              // global_addr[31:0]
    g0.w = (unsigned)((ga >> 32) & 0x01FFFFFFu)       // global_addr[56:32]
         | (2u << 30);                                // type = 2 ("image")
    i32x8 g1 = (i32x8)0;
    g1[0] = (int)(1u << 16);                          // data_size=1 -> 2 bytes
    g1[1] = (int)((tensor_x & 0xFFFFu) << 16);        // tensor_dim0[15:0]
    g1[2] = (int)((tensor_x >> 16) & 0xFFFFu)         // tensor_dim0[31:16]
          | (int)((tensor_y & 0xFFFFu) << 16);        // tensor_dim1[15:0]
    g1[3] = (int)((tensor_y >> 16) & 0xFFFFu)         // tensor_dim1[31:16]
          | (int)((tile_x & 0xFFFFu) << 16);          // tile_dim0
    g1[4] = (int)(tile_y & 0xFFFFu);                  // tile_dim1 (tile_dim2=0)
    g1[5] = (int)stride_x;                            // tensor_dim0_stride[31:0]
    i32x4 gz4 = (i32x4)0;
    i32x8 gz8 = (i32x8)0;
    __builtin_amdgcn_tensor_load_to_lds(g0, g1, gz4, gz4, gz8, 0);
#else
    (void)lds_off; (void)gaddr; (void)tile_x; (void)tile_y;
    (void)tensor_x; (void)tensor_y; (void)stride_x;
#endif
}

__device__ __forceinline__ unsigned lds_off_of(const void* p) {
    return (unsigned)(uintptr_t)p;                    // low 32 bits = LDS offset
}

// ---- WMMA fragment loaders (contiguous k = k0 + 16*half + s, wave32) -------------------
__device__ __forceinline__ v16bf load_frag_a(const u16* base, int lda, int k0, int lane) {
    const int half = lane >> 4;
    const u16* row = base + (size_t)(lane & 15) * lda + k0 + 16 * half;
    union { v16bf v; u32x4 q[2]; } u;
    u.q[0] = *(const u32x4*)(row);
    u.q[1] = *(const u32x4*)(row + 8);
    return u.v;
}

__device__ __forceinline__ v16bf load_frag_b(const u16* base, int ldb, int n0, int k0, int lane) {
    const u16* row = base + (size_t)(n0 + (lane & 15)) * ldb + k0 + 16 * (lane >> 4);
    union { v16bf v; u32x4 q[2]; } u;
    u.q[0] = *(const u32x4*)(row);
    u.q[1] = *(const u32x4*)(row + 8);
    return u.v;
}

__device__ __forceinline__ v8f wmma_bf16(v16bf a, v16bf b, v8f c) {
    return __builtin_amdgcn_wmma_f32_16x16x32_bf16(false, a, false, b,
                                                   (short)0, c, false, false);
}

// ---------------------------------------------------------------------------
// fp32 -> bf16 conversion (4-wide)
// ---------------------------------------------------------------------------
__global__ void conv_f32_bf16(const float* __restrict__ src,
                              u16* __restrict__ dst, int n) {
    int i = (blockIdx.x * blockDim.x + threadIdx.x) * 4;
    if (i < n) {
        float4 f = *(const float4*)(src + i);
        dst[i + 0] = f2bf(f.x);
        dst[i + 1] = f2bf(f.y);
        dst[i + 2] = f2bf(f.z);
        dst[i + 3] = f2bf(f.w);
    }
}

// ---------------------------------------------------------------------------
// Generic GEMM:  C[M,N] = A[M,K] * Bt[N,K]^T  (+bias) (+relu)
// Block = 256 threads = 8 waves; block tile 128(M) x 64(N); wave tile 32x32.
// K loop in 64-wide slices: A staged by per-lane async DMA, B (weights)
// staged by one TDM descriptor from wave 0.  Double-buffered LDS.
// ---------------------------------------------------------------------------
__global__ __launch_bounds__(256) void gemm_bt_wmma(
        const u16* __restrict__ A, const u16* __restrict__ Bt,
        const float* __restrict__ bias,
        float* __restrict__ Cf, u16* __restrict__ Cb,
        int M, int N, int K,
        int relu, int transOut, int Sdim, int Edim) {
    __shared__ u16 As[2][128][64];     // 16 KB x2
    __shared__ u16 Bs[2][64][64];      // 8 KB x2

    const int tid  = threadIdx.x;
    const int lane = tid & 31;
    const int w    = tid >> 5;              // wave id 0..7
    const int bmL  = (w >> 1) * 32;         // wave M offset within block tile
    const int bnL  = (w & 1) * 32;          // wave N offset within block tile
    const int bm   = blockIdx.y * 128 + bmL;
    const int bn   = blockIdx.x * 64  + bnL;
    const int hi   = lane >> 4;
    const int ln   = lane & 15;

    const u16* Ab = A  + (size_t)blockIdx.y * 128 * K;
    const u16* Bb = Bt + (size_t)blockIdx.x * 64  * K;

    v8f acc[2][2];
    for (int i = 0; i < 2; ++i)
        for (int j = 0; j < 2; ++j) acc[i][j] = v8f{};

    // stage one 64-wide K-slice
    auto stage = [&](int buf, int k0) {
        for (int c = tid; c < 1024; c += 256) {            // A: 128x64
            const int row = c >> 3, col = (c & 7) * 8;
            cp_async16(Ab + (size_t)row * K + k0 + col, &As[buf][row][col]);
        }
        if (HAVE_TDM) {
            if (w == 0)                                     // B: 64x64 via TDM
                tdm_load_2d(lds_off_of(&Bs[buf][0][0]), Bb + k0,
                            64, 64, (unsigned)K, (unsigned)N, (unsigned)K);
        } else {
            for (int c = tid; c < 512; c += 256) {          // fallback async
                const int row = c >> 3, col = (c & 7) * 8;
                cp_async16(Bb + (size_t)row * K + k0 + col, &Bs[buf][row][col]);
            }
        }
    };

    stage(0, 0);
    for (int k0 = 0; k0 < K; k0 += 64) {
        const int buf = (k0 >> 6) & 1;
        wait_async0();
        wait_tensor0();
        __syncthreads();
        if (k0 + 64 < K) stage(buf ^ 1, k0 + 64);

        for (int ks = 0; ks < 64; ks += 32) {
            v16bf a0 = load_frag_a(&As[buf][bmL][0],      64, ks, lane);
            v16bf a1 = load_frag_a(&As[buf][bmL + 16][0], 64, ks, lane);
            v16bf b0 = load_frag_b(&Bs[buf][0][0], 64, bnL,      ks, lane);
            v16bf b1 = load_frag_b(&Bs[buf][0][0], 64, bnL + 16, ks, lane);
            acc[0][0] = wmma_bf16(a0, b0, acc[0][0]);
            acc[0][1] = wmma_bf16(a0, b1, acc[0][1]);
            acc[1][0] = wmma_bf16(a1, b0, acc[1][0]);
            acc[1][1] = wmma_bf16(a1, b1, acc[1][1]);
        }
    }

    for (int im = 0; im < 2; ++im) {
        for (int in = 0; in < 2; ++in) {
            const int col = bn + in * 16 + ln;
            float bv = bias ? bias[col] : 0.f;
            for (int j = 0; j < 8; ++j) {
                const int row = bm + im * 16 + j + 8 * hi;
                float v = acc[im][in][j] + bv;
                if (relu) v = fmaxf(v, 0.f);
                if (Cf) Cf[(size_t)row * N + col] = v;
                if (Cb) {
                    if (transOut) {
                        const int bb = row / Sdim, ss = row % Sdim;
                        Cb[((size_t)bb * Edim + col) * Sdim + ss] = f2bf(v);
                    } else {
                        Cb[(size_t)row * N + col] = f2bf(v);
                    }
                }
            }
        }
    }
}

// ---------------------------------------------------------------------------
// Flash attention (no scale, no mask, faithful to reference).
// One block = 16 query rows of one batch.  8 waves.  Key blocks of 128.
// Phase 1: q staged by async DMA, k tiles by TDM; 64-wide K slices.
// Online softmax in LDS; probabilities requantized to bf16 in LDS, fed back
// through WMMA against v^T (stored [B,E,S]).  Wave w owns emb cols
// [w*96, w*96+96) -> 6 v8f accumulator tiles.
// ---------------------------------------------------------------------------
#define AKB 128

__global__ __launch_bounds__(256) void attn_flash_wmma(
        const u16* __restrict__ qb, const u16* __restrict__ kb,
        const u16* __restrict__ vT, u16* __restrict__ outb,
        int S, int E) {
    __shared__ float sc[16][AKB];        // fp32 score block (8 KB)
    __shared__ u16   pb[16][AKB];        // bf16 probabilities (4 KB)
    __shared__ u16   kt[2][AKB][64];     // staged k slices (16 KB x2)
    __shared__ u16   qt[2][16][64];      // staged q slices (2 KB x2)
    __shared__ float red[16][16];
    __shared__ float m_s[16], l_s[16], mnew_s[16], corr_s[16];

    const int tid  = threadIdx.x;
    const int lane = tid & 31;
    const int w    = tid >> 5;
    const int hi   = lane >> 4;
    const int ln   = lane & 15;
    const int b    = blockIdx.y;
    const int s0   = blockIdx.x * 16;

    const u16* qbase = qb + ((size_t)b * S + s0) * E;   // 16 q rows, lda=E
    const u16* kbase = kb + (size_t)b * S * E;          // [S,E]
    const u16* vbase = vT + (size_t)b * E * S;          // [E,S]

    if (tid < 16) { m_s[tid] = -1e30f; l_s[tid] = 0.f; }
    __syncthreads();

    v8f oacc[6];
    for (int i = 0; i < 6; ++i) oacc[i] = v8f{};

    for (int kb0 = 0; kb0 < S; kb0 += AKB) {
        // -- phase 1: score tile (16 x 128), TDM k + async q staging -------
        {
            auto stageQK = [&](int buf, int k0) {
                if (HAVE_TDM) {
                    if (w == 0)                              // k: 128x64 tile
                        tdm_load_2d(lds_off_of(&kt[buf][0][0]),
                                    kbase + (size_t)kb0 * E + k0,
                                    64, AKB, (unsigned)E, (unsigned)S,
                                    (unsigned)E);
                } else {
                    for (int c = tid; c < 1024; c += 256) {
                        const int row = c >> 3, col = (c & 7) * 8;
                        cp_async16(kbase + (size_t)(kb0 + row) * E + k0 + col,
                                   &kt[buf][row][col]);
                    }
                }
                if (tid < 128) {                             // q: 16x64
                    const int row = tid >> 3, col = (tid & 7) * 8;
                    cp_async16(qbase + (size_t)row * E + k0 + col,
                               &qt[buf][row][col]);
                }
            };

            v8f acc = v8f{};
            stageQK(0, 0);
            for (int k0 = 0; k0 < E; k0 += 64) {
                const int buf = (k0 >> 6) & 1;
                wait_async0();
                wait_tensor0();
                __syncthreads();
                if (k0 + 64 < E) stageQK(buf ^ 1, k0 + 64);
                for (int ks = 0; ks < 64; ks += 32) {
                    v16bf a  = load_frag_a(&qt[buf][0][0], 64, ks, lane);
                    v16bf bf = load_frag_b(&kt[buf][0][0], 64, w * 16, ks, lane);
                    acc = wmma_bf16(a, bf, acc);
                }
            }
            for (int j = 0; j < 8; ++j)
                sc[j + 8 * hi][w * 16 + ln] = acc[j];
        }
        __syncthreads();

        // -- phase 2: online softmax over this key block -------------------
        {
            const int r = tid >> 4, cs = tid & 15;     // row, 8-col segment
            float lmax = -1e30f;
            for (int c = 0; c < 8; ++c) lmax = fmaxf(lmax, sc[r][cs * 8 + c]);
            red[r][cs] = lmax;
            __syncthreads();
            if (cs == 0) {
                float mb = m_s[r];
                for (int j = 0; j < 16; ++j) mb = fmaxf(mb, red[r][j]);
                mnew_s[r] = mb;
                corr_s[r] = __expf(m_s[r] - mb);
            }
            __syncthreads();
            const float mn = mnew_s[r];
            float psum = 0.f;
            for (int c = 0; c < 8; ++c) {
                float p = __expf(sc[r][cs * 8 + c] - mn);
                pb[r][cs * 8 + c] = f2bf(p);
                psum += p;
            }
            red[r][cs] = psum;
            __syncthreads();
            if (cs == 0) {
                float sum = 0.f;
                for (int j = 0; j < 16; ++j) sum += red[r][j];
                l_s[r] = l_s[r] * corr_s[r] + sum;
                m_s[r] = mnew_s[r];
            }
            __syncthreads();
        }

        // -- phase 3: rescale accumulators, add p @ v ----------------------
        {
            for (int i = 0; i < 6; ++i) {
                v8f t = oacc[i];
                for (int j = 0; j < 8; ++j) t[j] *= corr_s[j + 8 * hi];
                oacc[i] = t;
            }
            for (int kk = 0; kk < AKB; kk += 32) {
                v16bf a = load_frag_a(&pb[0][0], AKB, kk, lane);  // LDS source
                if (kk + 32 < AKB)
                    __builtin_prefetch(vbase + (size_t)(w * 96 + ln) * S
                                             + kb0 + kk + 32, 0, 3);
                for (int i = 0; i < 6; ++i) {
                    const int n0 = w * 96 + i * 16;
                    v16bf bf = load_frag_b(vbase, S, n0, kb0 + kk, lane);
                    oacc[i] = wmma_bf16(a, bf, oacc[i]);
                }
            }
        }
        __syncthreads();
    }

    // -- finalize: divide by row sum, store bf16 ---------------------------
    for (int i = 0; i < 6; ++i) {
        const int n0 = w * 96 + i * 16;
        for (int j = 0; j < 8; ++j) {
            const int row = j + 8 * hi;
            float v = oacc[i][j] / l_s[row];
            outb[((size_t)b * S + s0 + row) * E + n0 + ln] = f2bf(v);
        }
    }
}

// ---------------------------------------------------------------------------
// Host-side orchestration
// ---------------------------------------------------------------------------
static inline void conv_launch(const float* s, u16* d, size_t n, hipStream_t st) {
    int blocks = (int)((n / 4 + 255) / 256);
    conv_f32_bf16<<<blocks, 256, 0, st>>>(s, d, (int)n);
}

extern "C" void kernel_launch(void* const* d_in, const int* in_sizes, int n_in,
                              void* d_out, int out_size, void* d_ws, size_t ws_size,
                              hipStream_t stream) {
    (void)in_sizes; (void)n_in; (void)out_size; (void)ws_size;
    const int Bb = 4, S = 4096, E = 768, F = 3072;
    const size_t nx = (size_t)Bb * S * E;          // 12,582,912
    const size_t M  = (size_t)Bb * S;              // 16384 rows

    const float* x  = (const float*)d_in[0];
    const float* Wq = (const float*)d_in[1];
    const float* Wk = (const float*)d_in[2];
    const float* Wv = (const float*)d_in[3];
    const float* W1 = (const float*)d_in[4];
    const float* b1 = (const float*)d_in[5];
    const float* W2 = (const float*)d_in[6];
    const float* b2 = (const float*)d_in[7];

    // workspace layout (bf16 elements), ~239 MB total
    u16* xb  = (u16*)d_ws;
    u16* Wqb = xb  + nx;
    u16* Wkb = Wqb + (size_t)E * E;
    u16* Wvb = Wkb + (size_t)E * E;
    u16* W1b = Wvb + (size_t)E * E;
    u16* W2b = W1b + (size_t)F * E;
    u16* qbf = W2b + (size_t)F * E;
    u16* kbf = qbf + nx;
    u16* vT  = kbf + nx;
    u16* nv  = vT  + nx;
    u16* hb  = nv  + nx;                            // [M, F] bf16

    // 1) convert fp32 -> bf16
    conv_launch(x,  xb,  nx, stream);
    conv_launch(Wq, Wqb, (size_t)E * E, stream);
    conv_launch(Wk, Wkb, (size_t)E * E, stream);
    conv_launch(Wv, Wvb, (size_t)E * E, stream);
    conv_launch(W1, W1b, (size_t)F * E, stream);
    conv_launch(W2, W2b, (size_t)F * E, stream);

    // 2) QKV projections (x @ W^T), v stored transposed [B,E,S]
    dim3 gq(E / 64, (int)(M / 128));
    gemm_bt_wmma<<<gq, 256, 0, stream>>>(xb, Wqb, nullptr, nullptr, qbf,
                                         (int)M, E, E, 0, 0, S, E);
    gemm_bt_wmma<<<gq, 256, 0, stream>>>(xb, Wkb, nullptr, nullptr, kbf,
                                         (int)M, E, E, 0, 0, S, E);
    gemm_bt_wmma<<<gq, 256, 0, stream>>>(xb, Wvb, nullptr, nullptr, vT,
                                         (int)M, E, E, 0, 1, S, E);

    // 3) flash attention -> nv (bf16 [B,S,E])
    dim3 ga(S / 16, Bb);
    attn_flash_wmma<<<ga, 256, 0, stream>>>(qbf, kbf, vT, nv, S, E);

    // 4) FFN1: h = relu(nv @ W1^T + b1)   [M, F] bf16
    dim3 g1(F / 64, (int)(M / 128));
    gemm_bt_wmma<<<g1, 256, 0, stream>>>(nv, W1b, b1, nullptr, hb,
                                         (int)M, F, E, 1, 0, S, E);

    // 5) FFN2: out = h @ W2^T + b2        [M, E] fp32 -> d_out
    dim3 g2(E / 64, (int)(M / 128));
    gemm_bt_wmma<<<g2, 256, 0, stream>>>(hb, W2b, b2, (float*)d_out, nullptr,
                                         (int)M, E, F, 0, 0, S, E);
}
